// Sent2GraphMatching_68959994904716
// MI455X (gfx1250) — compile-verified, matrix operand order checked by reading, not run on hardware
//
#include <hip/hip_runtime.h>
#include <hip/hip_bf16.h>
#include <math.h>

// ---------------------------------------------------------------------------
// Sent2Graph matching network for MI455X (gfx1250, wave32, WMMA).
//
// All heavy GEMMs run through a bf16 WMMA kernel (v_wmma_f32_16x16x32_bf16).
// A-operands are converted f32->bf16 on the fly while staging to LDS (the
// 660MB/330MB edge-feature inputs are streamed exactly twice each, once per
// 256-wide N block). Weights are pre-converted to zero-padded bf16 [N,Kp].
// Epilogue fuses bias + BatchNorm (+ReLU) (+second BatchNorm) and can emit
// f32 and/or bf16 outputs so downstream GEMMs consume bf16 directly.
// ---------------------------------------------------------------------------

typedef __bf16 bf16_t;
typedef __attribute__((ext_vector_type(16))) __bf16 v16bf;
typedef __attribute__((ext_vector_type(8)))  float  v8f;

union WFrag { v16bf v; uint4 q[2]; };

struct BnP { const float *g, *b, *m, *v; };

#define BB   32
#define KK   36
#define BK_  1152          // B*K
#define EE_  40320         // B*K*(K-1)
#define JJ   512

// ---------------------------------------------------------------------------
// WMMA GEMM:  C[M,N] = epilogue( A[M,K] * W[N,Kp]^T + bias )
// Block tile 128x256, BK=32, 256 threads = 8 waves as 2(M) x 4(N),
// wave tile 64x64 = 4x4 wmma 16x16 tiles.
// ---------------------------------------------------------------------------
template <typename TA>
__global__ __launch_bounds__(256)
void wmma_gemm_kernel(const TA* __restrict__ A, int lda, int M, int K, int Kp,
                      const bf16_t* __restrict__ W, int N,
                      const float* __restrict__ bias,
                      const float* __restrict__ g1, const float* __restrict__ b1,
                      const float* __restrict__ m1, const float* __restrict__ v1,
                      int relu,
                      const float* __restrict__ g2, const float* __restrict__ b2,
                      const float* __restrict__ m2, const float* __restrict__ v2,
                      float* __restrict__ outF, int ldF,
                      bf16_t* __restrict__ outB, int ldB)
{
    __shared__ __align__(16) bf16_t sA[128 * 32];
    __shared__ __align__(16) bf16_t sB[256 * 32];

    const int tid  = threadIdx.x;
    const int m0   = blockIdx.y * 128;
    const int n0   = blockIdx.x * 256;
    const int lane = tid & 31;
    const int wid  = tid >> 5;
    const int wm   = wid >> 2;        // 0..1
    const int wn   = wid & 3;         // 0..3
    const int lr   = lane & 15;
    const int lh   = lane >> 4;       // lane half (K-half select)

    v8f acc[4][4] = {};

    for (int k0 = 0; k0 < Kp; k0 += 32) {
        // ---- stage A tile (convert to bf16, zero-pad rows/cols) ----
        {
            const int r  = tid >> 1;
            const int cs = (tid & 1) << 4;
            const int gr = m0 + r;
            bf16_t* d = sA + r * 32 + cs;
            if (gr < M) {
                const TA* src = A + (long)gr * lda + k0 + cs;
                if (k0 + 32 < Kp)
                    __builtin_prefetch((const void*)(A + (long)gr * lda + k0 + 32 + cs), 0, 0);
                #pragma unroll
                for (int c = 0; c < 16; ++c) {
                    const int col = k0 + cs + c;
                    d[c] = (col < K) ? (bf16_t)(float)src[c] : (bf16_t)0.0f;
                }
            } else {
                #pragma unroll
                for (int c = 0; c < 16; ++c) d[c] = (bf16_t)0.0f;
            }
        }
        // ---- stage B tile (weights already bf16, [N,Kp], Kp%32==0) ----
        {
            const int n = n0 + tid;
            uint4* d = (uint4*)(sB + tid * 32);
            if (n < N) {
                const uint4* s = (const uint4*)(W + (long)n * Kp + k0);
                d[0] = s[0]; d[1] = s[1]; d[2] = s[2]; d[3] = s[3];
            } else {
                uint4 z = {0u, 0u, 0u, 0u};
                d[0] = z; d[1] = z; d[2] = z; d[3] = z;
            }
        }
        __syncthreads();

        // ---- build fragments per ISA 16-bit A-layout (B symmetric) ----
        WFrag af[4], bfr[4];
        #pragma unroll
        for (int mi = 0; mi < 4; ++mi) {
            const uint4* p = (const uint4*)(sA + (wm * 64 + mi * 16 + lr) * 32 + lh * 8);
            af[mi].q[0] = p[0];      // K 0..7  (or 8..15)
            af[mi].q[1] = p[2];      // K 16..23 (or 24..31)
        }
        #pragma unroll
        for (int ni = 0; ni < 4; ++ni) {
            const uint4* p = (const uint4*)(sB + (wn * 64 + ni * 16 + lr) * 32 + lh * 8);
            bfr[ni].q[0] = p[0];
            bfr[ni].q[1] = p[2];
        }
        #pragma unroll
        for (int mi = 0; mi < 4; ++mi)
            #pragma unroll
            for (int ni = 0; ni < 4; ++ni)
                acc[mi][ni] = __builtin_amdgcn_wmma_f32_16x16x32_bf16(
                    false, af[mi].v, false, bfr[ni].v, (short)0, acc[mi][ni],
                    false, false);
        __syncthreads();
    }

    // ---- epilogue: bias -> BN1 -> ReLU -> BN2, write f32/bf16 ----
    #pragma unroll
    for (int ni = 0; ni < 4; ++ni) {
        const int col = n0 + wn * 64 + ni * 16 + lr;
        if (col >= N) continue;
        const float bs = bias ? bias[col] : 0.0f;
        float sc1 = 1.0f, sh1 = 0.0f, sc2 = 1.0f, sh2 = 0.0f;
        if (g1) {
            const float iv = rsqrtf(v1[col] + 1e-5f);
            sc1 = g1[col] * iv;
            sh1 = b1[col] - m1[col] * sc1;
        }
        if (g2) {
            const float iv = rsqrtf(v2[col] + 1e-5f);
            sc2 = g2[col] * iv;
            sh2 = b2[col] - m2[col] * sc2;
        }
        #pragma unroll
        for (int mi = 0; mi < 4; ++mi) {
            #pragma unroll
            for (int e = 0; e < 8; ++e) {
                const int row = m0 + wm * 64 + mi * 16 + e + lh * 8;
                if (row >= M) continue;
                float x = acc[mi][ni][e] + bs;
                if (g1)  x = x * sc1 + sh1;
                if (relu) x = fmaxf(x, 0.0f);
                if (g2)  x = x * sc2 + sh2;
                if (outF) outF[(long)row * ldF + col] = x;
                if (outB) outB[(long)row * ldB + col] = (bf16_t)x;
            }
        }
    }
}

// ---------------------------------------------------------------------------
// Small support kernels (wave32 shuffle reductions etc.)
// ---------------------------------------------------------------------------
__global__ void cvt_weight_kernel(const float* __restrict__ src, bf16_t* __restrict__ dst,
                                  int N, int K, int Kp)
{
    long i = (long)blockIdx.x * blockDim.x + threadIdx.x;
    if (i >= (long)N * Kp) return;
    const int  c = (int)(i % Kp);
    const long n = i / Kp;
    dst[i] = (c < K) ? (bf16_t)src[n * K + c] : (bf16_t)0.0f;
}

__global__ void cvt_f2b_kernel(const float* __restrict__ src, bf16_t* __restrict__ dst, long n)
{
    long i = (long)blockIdx.x * blockDim.x + threadIdx.x;
    if (i < n) dst[i] = (bf16_t)src[i];
}

// e[r] = sum_d tanh(lang[b,d] + feat[r,d]) * w[d] + wb ; one wave per row
template <typename T>
__global__ void attn_score_kernel(const float* __restrict__ lang, const T* __restrict__ feat,
                                  const float* __restrict__ w, const float* __restrict__ wb,
                                  float* __restrict__ e, int rowsPerB)
{
    const int r = blockIdx.x;
    const int b = r / rowsPerB;
    const float* L = lang + b * JJ;
    const T* f = feat + (long)r * JJ;
    float s = 0.0f;
    for (int d = threadIdx.x; d < JJ; d += 32)
        s += tanhf(L[d] + (float)f[d]) * w[d];
    #pragma unroll
    for (int o = 16; o > 0; o >>= 1) s += __shfl_xor(s, o, 32);
    if (threadIdx.x == 0) e[r] = s + wb[0];
}

__global__ void softmax_kernel(float* __restrict__ e, int W)
{
    __shared__ float buf[64];
    float* p = e + (long)blockIdx.x * W;
    const int t = threadIdx.x;
    float m = -1e30f;
    for (int j = t; j < W; j += 64) m = fmaxf(m, p[j]);
    buf[t] = m; __syncthreads();
    for (int s = 32; s > 0; s >>= 1) { if (t < s) buf[t] = fmaxf(buf[t], buf[t + s]); __syncthreads(); }
    m = buf[0]; __syncthreads();
    float sum = 0.0f;
    for (int j = t; j < W; j += 64) { float x = __expf(p[j] - m); p[j] = x; sum += x; }
    buf[t] = sum; __syncthreads();
    for (int s = 32; s > 0; s >>= 1) { if (t < s) buf[t] += buf[t + s]; __syncthreads(); }
    const float inv = 1.0f / buf[0];
    __syncthreads();
    for (int j = t; j < W; j += 64) p[j] *= inv;
}

// dst[r, 0:width] = bf16( src[r*lds + c] * a[r] )
__global__ void scale_bf16_kernel(const float* __restrict__ src, int lds_, int width,
                                  const float* __restrict__ a, bf16_t* __restrict__ dst, long rows)
{
    long i = (long)blockIdx.x * blockDim.x + threadIdx.x;
    if (i >= rows * width) return;
    const long r = i / width;
    const int  c = (int)(i % width);
    dst[r * width + c] = (bf16_t)(src[r * lds_ + c] * a[r]);
}

// masked-scatter replacement: A[b,i,c] = (c==i) ? 0 : a[b,i, c - (c>i)]
__global__ void build_adj_kernel(const float* __restrict__ a, float* __restrict__ Adj)
{
    const int i = blockIdx.x * blockDim.x + threadIdx.x;
    if (i >= BB * KK * KK) return;
    const int c   = i % KK;
    const int bi  = i / KK;
    const int row = bi % KK;
    const int b   = bi / KK;
    float v = 0.0f;
    if (c != row) {
        const int j = c - (c > row ? 1 : 0);
        v = a[((long)b * KK + row) * (KK - 1) + j];
    }
    Adj[i] = v;
}

// H[r,d] = sum_j S[r,j,d] * a[r,j]   (S bf16, per-(b,i) edge aggregate)
__global__ void edge_aggregate_kernel(const bf16_t* __restrict__ S, const float* __restrict__ a,
                                      float* __restrict__ H)
{
    long i = (long)blockIdx.x * blockDim.x + threadIdx.x;
    if (i >= (long)BK_ * JJ) return;
    const long r = i / JJ;
    const int  d = (int)(i % JJ);
    const bf16_t* s = S + r * (long)(KK - 1) * JJ + d;
    const float* av = a + r * (KK - 1);
    float acc = 0.0f;
    for (int j = 0; j < KK - 1; ++j) acc += (float)s[(long)j * JJ] * av[j];
    H[i] = acc;
}

// Xb[b,i,d] = bf16( tanh( sum_j Adj[b,i,j] * H[b,j,d] ) )
__global__ void msg_pass_kernel(const float* __restrict__ Adj, const float* __restrict__ H,
                                bf16_t* __restrict__ Xb)
{
    long i = (long)blockIdx.x * blockDim.x + threadIdx.x;
    if (i >= (long)BK_ * JJ) return;
    const int  d  = (int)(i % JJ);
    const long rk = i / JJ;
    const int  k  = (int)(rk % KK);
    const int  b  = (int)(rk / KK);
    const float* Ar = Adj + ((long)b * KK + k) * KK;
    const float* Hb = H + (long)b * KK * JJ + d;
    float s = 0.0f;
    for (int j = 0; j < KK; ++j) s += Ar[j] * Hb[(long)j * JJ];
    Xb[i] = (bf16_t)tanhf(s);
}

__global__ void gru_combine_kernel(const float* __restrict__ gi, const float* __restrict__ gh,
                                   float* __restrict__ h)
{
    long i = (long)blockIdx.x * blockDim.x + threadIdx.x;
    if (i >= (long)BK_ * JJ) return;
    const long r = i / JJ;
    const int  d = (int)(i % JJ);
    const float* gir = gi + r * (3 * JJ);
    const float* ghr = gh + r * (3 * JJ);
    const float ir = gir[d],        hr = ghr[d];
    const float iz = gir[JJ + d],   hz = ghr[JJ + d];
    const float in_ = gir[2*JJ + d], hn = ghr[2*JJ + d];
    const float rg = 1.0f / (1.0f + __expf(-(ir + hr)));
    const float z  = 1.0f / (1.0f + __expf(-(iz + hz)));
    const float n  = tanhf(in_ + rg * hn);
    h[i] = (1.0f - z) * n + z * h[i];
}

// cosine score: one wave per (b,k) row
__global__ void cos_score_kernel(const float* __restrict__ sent, const float* __restrict__ xm,
                                 float* __restrict__ s)
{
    const int r = blockIdx.x;
    const int b = r / KK;
    const float* sv = sent + b * JJ;
    const float* xv = xm + (long)r * JJ;
    float d = 0.0f, ns = 0.0f, nx = 0.0f;
    for (int i = threadIdx.x; i < JJ; i += 32) {
        const float a = sv[i], c = xv[i];
        d += a * c; ns += a * a; nx += c * c;
    }
    #pragma unroll
    for (int o = 16; o > 0; o >>= 1) {
        d  += __shfl_xor(d,  o, 32);
        ns += __shfl_xor(ns, o, 32);
        nx += __shfl_xor(nx, o, 32);
    }
    if (threadIdx.x == 0)
        s[r] = d / ((sqrtf(ns) + 1e-8f) * (sqrtf(nx) + 1e-8f));
}

__global__ void final_combine_kernel(const float* __restrict__ sc, const float* __restrict__ wts,
                                     float* __restrict__ out)
{
    const int i = blockIdx.x * blockDim.x + threadIdx.x;
    if (i >= BB * KK) return;
    const int b = i / KK;
    out[i] = wts[b * 4 + 0] * sc[i]
           + wts[b * 4 + 1] * sc[BB * KK + i]
           + wts[b * 4 + 2] * sc[2 * BB * KK + i]
           + wts[b * 4 + 3] * sc[3 * BB * KK + i];
}

// ---------------------------------------------------------------------------
// Host-side launch helpers
// ---------------------------------------------------------------------------
static inline void gemm_f32(hipStream_t st, const float* A, int lda, int M, int K, int Kp,
                            const bf16_t* W, int N, const float* bias, BnP bn1, int relu, BnP bn2,
                            float* oF, int ldF, bf16_t* oB, int ldB)
{
    dim3 g((unsigned)((N + 255) / 256), (unsigned)((M + 127) / 128));
    wmma_gemm_kernel<float><<<g, 256, 0, st>>>(A, lda, M, K, Kp, W, N, bias,
        bn1.g, bn1.b, bn1.m, bn1.v, relu, bn2.g, bn2.b, bn2.m, bn2.v, oF, ldF, oB, ldB);
}
static inline void gemm_b16(hipStream_t st, const bf16_t* A, int lda, int M, int K, int Kp,
                            const bf16_t* W, int N, const float* bias, BnP bn1, int relu, BnP bn2,
                            float* oF, int ldF, bf16_t* oB, int ldB)
{
    dim3 g((unsigned)((N + 255) / 256), (unsigned)((M + 127) / 128));
    wmma_gemm_kernel<bf16_t><<<g, 256, 0, st>>>(A, lda, M, K, Kp, W, N, bias,
        bn1.g, bn1.b, bn1.m, bn1.v, relu, bn2.g, bn2.b, bn2.m, bn2.v, oF, ldF, oB, ldB);
}

// Flattened input indices (setup_inputs() dict order, params flattened
// recursively in insertion order: lin{W,b}, bn{g,b,m,v}, emb{l1,bn1,l2,bn2,bn3},
// gru{Wi,bi,Wh,bh}).
enum {
    IN_VIS = 0, IN_LOC, IN_SAME, IN_LRL,
    IN_LVIS, IN_LORD, IN_LSAME, IN_LLOC,
    IN_WTS, IN_NOBJ, IN_S2I,
    B_VIS = 11, B_LOC = 27, B_SAME = 43, B_LRL = 59,
    I_LANGVIS_W = 75, I_LANGLOC_W = 77, I_LANGSAME_W = 79, I_LANGORD_W = 81,
    I_NODEVIS_W = 83, I_NODEORD_W = 85, I_EDGESAME_W = 87, I_EDGELOC_W = 89,
    I_ATTNVIS_W = 91, I_ATTNORD_W = 93, I_ATTNSAME_W = 95, I_ATTNLOC_W = 97,
    I_MLVIS_W = 99, I_MLORD_W = 101, I_MLSAME_W = 103, I_MLLOC_W = 105,
    I_MVIS_W = 107, I_MORD_W = 109, I_MSAME_W = 111, I_MLOC_W = 113,
    I_SGRU_WI = 115, I_SGRU_BI = 116, I_SGRU_WH = 117, I_SGRU_BH = 118,
    I_LGRU_WI = 119, I_LGRU_BI = 120, I_LGRU_WH = 121, I_LGRU_BH = 122
};

extern "C" void kernel_launch(void* const* d_in, const int* in_sizes, int n_in,
                              void* d_out, int out_size, void* d_ws, size_t ws_size,
                              hipStream_t stream)
{
    auto F = [&](int i) { return (const float*)d_in[i]; };
    auto bnp = [&](int ib) { return BnP{F(ib), F(ib + 1), F(ib + 2), F(ib + 3)}; };
    const BnP BN0{nullptr, nullptr, nullptr, nullptr};

    // ---------------- workspace layout (bump allocator) ----------------
    uint8_t* ws = (uint8_t*)d_ws;
    size_t off = 0;
    auto alloc = [&](size_t bytes) -> void* {
        off = (off + 255) & ~(size_t)255;
        void* p = ws + off;
        off += bytes;
        return p;
    };

    float*  ORD   = (float*) alloc((size_t)BK_ * 1024 * 4);   // [SUB | locn] f32
    bf16_t* ORDB  = (bf16_t*)alloc((size_t)BK_ * 1024 * 2);
    bf16_t* HBF   = (bf16_t*)alloc((size_t)EE_ * JJ * 2);     // l1 hidden (reused)
    bf16_t* SAMEB = (bf16_t*)alloc((size_t)EE_ * JJ * 2);
    bf16_t* LOCEB = (bf16_t*)alloc((size_t)EE_ * JJ * 2);
    bf16_t* EEB   = (bf16_t*)alloc((size_t)EE_ * JJ * 2);     // edge linear out (reused)
    float*  NE    = (float*) alloc((size_t)BK_ * JJ * 4);
    float*  EATT  = (float*) alloc((size_t)EE_ * 4);
    float*  ANODE = (float*) alloc((size_t)BK_ * 4);
    float*  ADJS  = (float*) alloc((size_t)BB * KK * KK * 4);
    float*  ADJL  = (float*) alloc((size_t)BB * KK * KK * 4);
    float*  HS    = (float*) alloc((size_t)BK_ * JJ * 4);
    float*  HL    = (float*) alloc((size_t)BK_ * JJ * 4);
    bf16_t* XB    = (bf16_t*)alloc((size_t)BK_ * JJ * 2);
    bf16_t* HB2   = (bf16_t*)alloc((size_t)BK_ * JJ * 2);
    float*  GI    = (float*) alloc((size_t)BK_ * 1536 * 4);
    float*  GH    = (float*) alloc((size_t)BK_ * 1536 * 4);
    bf16_t* SUBB  = (bf16_t*)alloc((size_t)BK_ * JJ * 2);
    bf16_t* ORDIB = (bf16_t*)alloc((size_t)BK_ * 1024 * 2);
    float*  XM    = (float*) alloc((size_t)BK_ * JJ * 4);
    float*  SC    = (float*) alloc((size_t)4 * BK_ * 4);
    float*  LANG  = (float*) alloc((size_t)8 * BB * JJ * 4);  // 8 language slots

    bf16_t* Wvis1 = (bf16_t*)alloc((size_t)512 * 2048 * 2);
    bf16_t* Wvis2 = (bf16_t*)alloc((size_t)512 * 512 * 2);
    bf16_t* Wloc1 = (bf16_t*)alloc((size_t)512 * 32 * 2);
    bf16_t* Wloc2 = (bf16_t*)alloc((size_t)512 * 512 * 2);
    bf16_t* Wsam1 = (bf16_t*)alloc((size_t)512 * 4128 * 2);
    bf16_t* Wsam2 = (bf16_t*)alloc((size_t)512 * 512 * 2);
    bf16_t* Wlrl1 = (bf16_t*)alloc((size_t)512 * 2080 * 2);
    bf16_t* Wlrl2 = (bf16_t*)alloc((size_t)512 * 512 * 2);
    bf16_t* WlgV  = (bf16_t*)alloc((size_t)512 * 1024 * 2);
    bf16_t* WlgL  = (bf16_t*)alloc((size_t)512 * 1024 * 2);
    bf16_t* WlgS  = (bf16_t*)alloc((size_t)512 * 1024 * 2);
    bf16_t* WlgO  = (bf16_t*)alloc((size_t)512 * 1024 * 2);
    bf16_t* WmlV  = (bf16_t*)alloc((size_t)512 * 1024 * 2);
    bf16_t* WmlO  = (bf16_t*)alloc((size_t)512 * 1024 * 2);
    bf16_t* WmlS  = (bf16_t*)alloc((size_t)512 * 1024 * 2);
    bf16_t* WmlL  = (bf16_t*)alloc((size_t)512 * 1024 * 2);
    bf16_t* Wnv   = (bf16_t*)alloc((size_t)512 * 512 * 2);
    bf16_t* Wno   = (bf16_t*)alloc((size_t)512 * 1024 * 2);
    bf16_t* Wes   = (bf16_t*)alloc((size_t)512 * 512 * 2);
    bf16_t* Wel   = (bf16_t*)alloc((size_t)512 * 512 * 2);
    bf16_t* Wmv   = (bf16_t*)alloc((size_t)512 * 512 * 2);
    bf16_t* Wmo   = (bf16_t*)alloc((size_t)512 * 1024 * 2);
    bf16_t* Wms   = (bf16_t*)alloc((size_t)512 * 512 * 2);
    bf16_t* Wml   = (bf16_t*)alloc((size_t)512 * 512 * 2);
    bf16_t* Wsgi  = (bf16_t*)alloc((size_t)1536 * 512 * 2);
    bf16_t* Wsgh  = (bf16_t*)alloc((size_t)1536 * 512 * 2);
    bf16_t* Wlgi  = (bf16_t*)alloc((size_t)1536 * 512 * 2);
    bf16_t* Wlgh  = (bf16_t*)alloc((size_t)1536 * 512 * 2);

    if (off > ws_size) return;   // workspace too small; bail rather than corrupt

    auto cvtW = [&](int idx, bf16_t* dst, int N, int K, int Kp) {
        long tot = (long)N * Kp;
        cvt_weight_kernel<<<(unsigned)((tot + 255) / 256), 256, 0, stream>>>(F(idx), dst, N, K, Kp);
    };

    // ---------------- weight conversions (f32 -> padded bf16) ----------------
    cvtW(B_VIS + 0, Wvis1, 512, 2048, 2048);  cvtW(B_VIS + 6, Wvis2, 512, 512, 512);
    cvtW(B_LOC + 0, Wloc1, 512, 5, 32);       cvtW(B_LOC + 6, Wloc2, 512, 512, 512);
    cvtW(B_SAME + 0, Wsam1, 512, 4106, 4128); cvtW(B_SAME + 6, Wsam2, 512, 512, 512);
    cvtW(B_LRL + 0, Wlrl1, 512, 2058, 2080);  cvtW(B_LRL + 6, Wlrl2, 512, 512, 512);
    cvtW(I_LANGVIS_W, WlgV, 512, 1024, 1024); cvtW(I_LANGLOC_W, WlgL, 512, 1024, 1024);
    cvtW(I_LANGSAME_W, WlgS, 512, 1024, 1024);cvtW(I_LANGORD_W, WlgO, 512, 1024, 1024);
    cvtW(I_MLVIS_W, WmlV, 512, 1024, 1024);   cvtW(I_MLORD_W, WmlO, 512, 1024, 1024);
    cvtW(I_MLSAME_W, WmlS, 512, 1024, 1024);  cvtW(I_MLLOC_W, WmlL, 512, 1024, 1024);
    cvtW(I_NODEVIS_W, Wnv, 512, 512, 512);    cvtW(I_NODEORD_W, Wno, 512, 1024, 1024);
    cvtW(I_EDGESAME_W, Wes, 512, 512, 512);   cvtW(I_EDGELOC_W, Wel, 512, 512, 512);
    cvtW(I_MVIS_W, Wmv, 512, 512, 512);       cvtW(I_MORD_W, Wmo, 512, 1024, 1024);
    cvtW(I_MSAME_W, Wms, 512, 512, 512);      cvtW(I_MLOC_W, Wml, 512, 512, 512);
    cvtW(I_SGRU_WI, Wsgi, 1536, 512, 512);    cvtW(I_SGRU_WH, Wsgh, 1536, 512, 512);
    cvtW(I_LGRU_WI, Wlgi, 1536, 512, 512);    cvtW(I_LGRU_WH, Wlgh, 1536, 512, 512);

    // ---------------- embedding towers ----------------
    auto run_emb = [&](const float* X, int Kin, int Kp1, int M, int base,
                       bf16_t* W1, bf16_t* W2,
                       float* outF, int ldF, bf16_t* outB, int ldB) {
        // l1 + bn1 + relu -> HBF (bf16)
        gemm_f32(stream, X, Kin, M, Kin, Kp1, W1, 512, F(base + 1), bnp(base + 2), 1, BN0,
                 nullptr, 0, HBF, 512);
        // l2 + bn2 + relu + bn3
        gemm_b16(stream, HBF, 512, M, 512, 512, W2, 512, F(base + 7), bnp(base + 8), 1,
                 bnp(base + 12), outF, ldF, outB, ldB);
    };

    run_emb(F(IN_VIS), 2048, 2048, BK_, B_VIS, Wvis1, Wvis2, ORD,       1024, ORDB,       1024); // SUB
    run_emb(F(IN_LOC), 5,    32,   BK_, B_LOC, Wloc1, Wloc2, ORD + 512, 1024, ORDB + 512, 1024); // locn
    run_emb(F(IN_SAME), 4106, 4128, EE_, B_SAME, Wsam1, Wsam2, nullptr, 0, SAMEB, 512);          // SAME
    run_emb(F(IN_LRL),  2058, 2080, EE_, B_LRL,  Wlrl1, Wlrl2, nullptr, 0, LOCEB, 512);          // LOCE

    // ---------------- language linears (8x, M=32) ----------------
    float* Lvis  = LANG + 0 * BB * JJ; float* Lloc  = LANG + 1 * BB * JJ;
    float* Lsame = LANG + 2 * BB * JJ; float* Lord  = LANG + 3 * BB * JJ;
    float* MLvis = LANG + 4 * BB * JJ; float* MLord = LANG + 5 * BB * JJ;
    float* MLsam = LANG + 6 * BB * JJ; float* MLloc = LANG + 7 * BB * JJ;
    auto lin32 = [&](const float* X, bf16_t* W, const float* b, float* out) {
        gemm_f32(stream, X, 1024, BB, 1024, 1024, W, 512, b, BN0, 0, BN0, out, 512, nullptr, 0);
    };
    lin32(F(IN_LVIS),  WlgV, F(I_LANGVIS_W + 1),  Lvis);
    lin32(F(IN_LLOC),  WlgL, F(I_LANGLOC_W + 1),  Lloc);
    lin32(F(IN_LSAME), WlgS, F(I_LANGSAME_W + 1), Lsame);
    lin32(F(IN_LORD),  WlgO, F(I_LANGORD_W + 1),  Lord);
    lin32(F(IN_LVIS),  WmlV, F(I_MLVIS_W + 1),  MLvis);
    lin32(F(IN_LORD),  WmlO, F(I_MLORD_W + 1),  MLord);
    lin32(F(IN_LSAME), WmlS, F(I_MLSAME_W + 1), MLsam);
    lin32(F(IN_LLOC),  WmlL, F(I_MLLOC_W + 1),  MLloc);

    const int TB = 256;
    const unsigned gBKJ  = (unsigned)(((long)BK_ * JJ + TB - 1) / TB);
    const unsigned gBK1k = (unsigned)(((long)BK_ * 1024 + TB - 1) / TB);

    // ---------------- node attention (vis / ord) ----------------
    gemm_b16(stream, ORDB, 1024, BK_, 512, 512, Wnv, 512, F(I_NODEVIS_W + 1), BN0, 0, BN0,
             NE, 512, nullptr, 0);
    attn_score_kernel<float><<<BK_, 32, 0, stream>>>(Lvis, NE, F(I_ATTNVIS_W), F(I_ATTNVIS_W + 1), ANODE, KK);
    softmax_kernel<<<BB, 64, 0, stream>>>(ANODE, KK);
    scale_bf16_kernel<<<gBKJ, TB, 0, stream>>>(ORD, 1024, 512, ANODE, SUBB, BK_);   // sub (bf16)

    gemm_b16(stream, ORDB, 1024, BK_, 1024, 1024, Wno, 512, F(I_NODEORD_W + 1), BN0, 0, BN0,
             NE, 512, nullptr, 0);
    attn_score_kernel<float><<<BK_, 32, 0, stream>>>(Lord, NE, F(I_ATTNORD_W), F(I_ATTNORD_W + 1), ANODE, KK);
    softmax_kernel<<<BB, 64, 0, stream>>>(ANODE, KK);
    scale_bf16_kernel<<<gBK1k, TB, 0, stream>>>(ORD, 1024, 1024, ANODE, ORDIB, BK_); // ordi (bf16)

    // ---------------- edge attention: same then loc ----------------
    gemm_b16(stream, SAMEB, 512, EE_, 512, 512, Wes, 512, F(I_EDGESAME_W + 1), BN0, 0, BN0,
             nullptr, 0, EEB, 512);
    attn_score_kernel<bf16_t><<<EE_, 32, 0, stream>>>(Lsame, EEB, F(I_ATTNSAME_W), F(I_ATTNSAME_W + 1),
                                                      EATT, KK * (KK - 1));
    softmax_kernel<<<BK_, 64, 0, stream>>>(EATT, KK - 1);
    build_adj_kernel<<<(BB * KK * KK + TB - 1) / TB, TB, 0, stream>>>(EATT, ADJS);
    edge_aggregate_kernel<<<gBKJ, TB, 0, stream>>>(SAMEB, EATT, HS);

    gemm_b16(stream, LOCEB, 512, EE_, 512, 512, Wel, 512, F(I_EDGELOC_W + 1), BN0, 0, BN0,
             nullptr, 0, EEB, 512);
    attn_score_kernel<bf16_t><<<EE_, 32, 0, stream>>>(Lloc, EEB, F(I_ATTNLOC_W), F(I_ATTNLOC_W + 1),
                                                      EATT, KK * (KK - 1));
    softmax_kernel<<<BK_, 64, 0, stream>>>(EATT, KK - 1);
    build_adj_kernel<<<(BB * KK * KK + TB - 1) / TB, TB, 0, stream>>>(EATT, ADJL);
    edge_aggregate_kernel<<<gBKJ, TB, 0, stream>>>(LOCEB, EATT, HL);

    // ---------------- GRU message-passing rounds ----------------
    for (int rnd = 0; rnd < 3; ++rnd) {
        // same branch
        msg_pass_kernel<<<gBKJ, TB, 0, stream>>>(ADJS, HS, XB);
        cvt_f2b_kernel<<<gBKJ, TB, 0, stream>>>(HS, HB2, (long)BK_ * JJ);
        gemm_b16(stream, XB, 512, BK_, 512, 512, Wsgi, 1536, F(I_SGRU_BI), BN0, 0, BN0, GI, 1536, nullptr, 0);
        gemm_b16(stream, HB2, 512, BK_, 512, 512, Wsgh, 1536, F(I_SGRU_BH), BN0, 0, BN0, GH, 1536, nullptr, 0);
        gru_combine_kernel<<<gBKJ, TB, 0, stream>>>(GI, GH, HS);
        // loc branch
        msg_pass_kernel<<<gBKJ, TB, 0, stream>>>(ADJL, HL, XB);
        cvt_f2b_kernel<<<gBKJ, TB, 0, stream>>>(HL, HB2, (long)BK_ * JJ);
        gemm_b16(stream, XB, 512, BK_, 512, 512, Wlgi, 1536, F(I_LGRU_BI), BN0, 0, BN0, GI, 1536, nullptr, 0);
        gemm_b16(stream, HB2, 512, BK_, 512, 512, Wlgh, 1536, F(I_LGRU_BH), BN0, 0, BN0, GH, 1536, nullptr, 0);
        gru_combine_kernel<<<gBKJ, TB, 0, stream>>>(GI, GH, HL);
    }

    // ---------------- matching scores ----------------
    gemm_b16(stream, SUBB, 512, BK_, 512, 512, Wmv, 512, F(I_MVIS_W + 1), BN0, 0, BN0, XM, 512, nullptr, 0);
    cos_score_kernel<<<BK_, 32, 0, stream>>>(MLvis, XM, SC + 0 * BK_);

    gemm_b16(stream, ORDIB, 1024, BK_, 1024, 1024, Wmo, 512, F(I_MORD_W + 1), BN0, 0, BN0, XM, 512, nullptr, 0);
    cos_score_kernel<<<BK_, 32, 0, stream>>>(MLord, XM, SC + 1 * BK_);

    cvt_f2b_kernel<<<gBKJ, TB, 0, stream>>>(HS, HB2, (long)BK_ * JJ);
    gemm_b16(stream, HB2, 512, BK_, 512, 512, Wms, 512, F(I_MSAME_W + 1), BN0, 0, BN0, XM, 512, nullptr, 0);
    cos_score_kernel<<<BK_, 32, 0, stream>>>(MLsam, XM, SC + 2 * BK_);

    cvt_f2b_kernel<<<gBKJ, TB, 0, stream>>>(HL, HB2, (long)BK_ * JJ);
    gemm_b16(stream, HB2, 512, BK_, 512, 512, Wml, 512, F(I_MLOC_W + 1), BN0, 0, BN0, XM, 512, nullptr, 0);
    cos_score_kernel<<<BK_, 32, 0, stream>>>(MLloc, XM, SC + 3 * BK_);

    final_combine_kernel<<<(BB * KK + TB - 1) / TB, TB, 0, stream>>>(SC, F(IN_WTS), (float*)d_out);

    (void)in_sizes; (void)n_in; (void)out_size;
}
